// VectorQuantizer_61297773248851
// MI455X (gfx1250) — matrix operand order repaired and checked by reading, not compiled, not run on hardware
//
#include <hip/hip_runtime.h>
#include <hip/hip_bf16.h>

typedef __bf16 v16bf __attribute__((ext_vector_type(16)));
typedef float  v8f   __attribute__((ext_vector_type(8)));

#define N_CODES 1024
#define DIM     64
#define HW      4096      // H*W
#define CHW     262144    // C*H*W
#define NROWS   131072    // B*H*W
#define NPAIRS  4096      // NROWS/32 : one wave handles 32 rows (two 16-row A tiles)
#define NELEM   8388608   // B*C*H*W

// ---------------------------------------------------------------------------
// Prep: split codebook into bf16 hi/lo, compute ||e||^2, zero usage flags
// ---------------------------------------------------------------------------
__global__ __launch_bounds__(256) void vq_prep(const float* __restrict__ codebook,
                                               __bf16* __restrict__ cb_hi,
                                               __bf16* __restrict__ cb_lo,
                                               float* __restrict__ cbnorm,
                                               int* __restrict__ used) {
  int code = blockIdx.x * blockDim.x + threadIdx.x;
  if (code >= N_CODES) return;
  float nrm = 0.f;
  #pragma unroll 8
  for (int d = 0; d < DIM; ++d) {
    float x = codebook[code * DIM + d];
    __bf16 h = (__bf16)x;
    cb_hi[code * DIM + d] = h;
    cb_lo[code * DIM + d] = (__bf16)(x - (float)h);
    nrm += x * x;
  }
  cbnorm[code] = nrm;
  used[code] = 0;
}

// ---------------------------------------------------------------------------
// Main: fused (z . e) WMMA GEMM + argmin + gather + per-pair loss partial.
// One wave owns TWO adjacent 16-row A tiles (32 z rows) vs all 1024 codes:
// halves codebook L2 traffic, 12 WMMAs per B-fragment fetch per code tile.
// ---------------------------------------------------------------------------
__global__ __launch_bounds__(256) void vq_main(const float* __restrict__ z,
                                               const float* __restrict__ codebook,
                                               const __bf16* __restrict__ cb_hi,
                                               const __bf16* __restrict__ cb_lo,
                                               const float* __restrict__ cbnorm,
                                               float* __restrict__ zq_out,
                                               int* __restrict__ idx_out,
                                               int* __restrict__ used,
                                               float* __restrict__ partials) {
  __shared__ int sIdx[8][32];
  const int wave = threadIdx.x >> 5;
  const int lane = threadIdx.x & 31;
  const int g    = lane >> 4;     // lane half-group
  const int m    = lane & 15;     // row-in-tile (A) / code-in-tile (B)
  const int pair = blockIdx.x * 8 + wave;
  const int n0   = pair * 32;     // first z_flat row (w0 in {0,32}: never crosses W)
  const int b    = n0 >> 12;      // n0 / 4096
  const int hw0  = n0 & 4095;     // h*64 + w0

  // ---- Load A fragments (two row tiles) with hi/lo bf16 split ----
  // A-layout (16-bit 16x32): lane half g, vec elem j -> K = 16*g + j (matches B)
  const float* zb0 = z + (size_t)b * CHW + hw0 + m;        // rows n0 + m
  const float* zb1 = zb0 + 16;                             // rows n0 + 16 + m
  v16bf a0_hi[2], a0_lo[2], a1_hi[2], a1_lo[2];
  float sumsq0 = 0.f, sumsq1 = 0.f;
  #pragma unroll
  for (int kk = 0; kk < 2; ++kk) {
    #pragma unroll
    for (int j = 0; j < 16; ++j) {
      int d = kk * 32 + 16 * g + j;
      size_t off = (size_t)d * HW;
      float x0 = zb0[off];                    // 16-lane coalesced (64B lines)
      float x1 = zb1[off];
      sumsq0 += x0 * x0;
      sumsq1 += x1 * x1;
      __bf16 h0 = (__bf16)x0;
      __bf16 h1 = (__bf16)x1;
      a0_hi[kk][j] = h0;  a0_lo[kk][j] = (__bf16)(x0 - (float)h0);
      a1_hi[kk][j] = h1;  a1_lo[kk][j] = (__bf16)(x1 - (float)h1);
    }
  }

  // ---- Sweep all 1024 codes in 16-wide tiles ----
  float bv0[8], bv1[8]; int bi0[8], bi1[8];
  #pragma unroll
  for (int r = 0; r < 8; ++r) {
    bv0[r] = 3.4e38f; bi0[r] = 0;
    bv1[r] = 3.4e38f; bi1[r] = 0;
  }

  for (int t = 0; t < 64; ++t) {
    int code = t * 16 + m;
    // B order used here (contiguous, same as A): lane half g, elem j -> K = 16*g + j
    const __bf16* ch = cb_hi + (size_t)code * DIM + g * 16;
    const __bf16* cl = cb_lo + (size_t)code * DIM + g * 16;
    v16bf b_hi0 = *(const v16bf*)(ch);
    v16bf b_hi1 = *(const v16bf*)(ch + 32);
    v16bf b_lo0 = *(const v16bf*)(cl);
    v16bf b_lo1 = *(const v16bf*)(cl + 32);
    float cn = cbnorm[code];

    v8f acc0 = {}, acc1 = {};
    acc0 = __builtin_amdgcn_wmma_f32_16x16x32_bf16(false, a0_hi[0], false, b_hi0, (short)0, acc0, false, false);
    acc1 = __builtin_amdgcn_wmma_f32_16x16x32_bf16(false, a1_hi[0], false, b_hi0, (short)0, acc1, false, false);
    acc0 = __builtin_amdgcn_wmma_f32_16x16x32_bf16(false, a0_hi[1], false, b_hi1, (short)0, acc0, false, false);
    acc1 = __builtin_amdgcn_wmma_f32_16x16x32_bf16(false, a1_hi[1], false, b_hi1, (short)0, acc1, false, false);
    acc0 = __builtin_amdgcn_wmma_f32_16x16x32_bf16(false, a0_hi[0], false, b_lo0, (short)0, acc0, false, false);
    acc1 = __builtin_amdgcn_wmma_f32_16x16x32_bf16(false, a1_hi[0], false, b_lo0, (short)0, acc1, false, false);
    acc0 = __builtin_amdgcn_wmma_f32_16x16x32_bf16(false, a0_hi[1], false, b_lo1, (short)0, acc0, false, false);
    acc1 = __builtin_amdgcn_wmma_f32_16x16x32_bf16(false, a1_hi[1], false, b_lo1, (short)0, acc1, false, false);
    acc0 = __builtin_amdgcn_wmma_f32_16x16x32_bf16(false, a0_lo[0], false, b_hi0, (short)0, acc0, false, false);
    acc1 = __builtin_amdgcn_wmma_f32_16x16x32_bf16(false, a1_lo[0], false, b_hi0, (short)0, acc1, false, false);
    acc0 = __builtin_amdgcn_wmma_f32_16x16x32_bf16(false, a0_lo[1], false, b_hi1, (short)0, acc0, false, false);
    acc1 = __builtin_amdgcn_wmma_f32_16x16x32_bf16(false, a1_lo[1], false, b_hi1, (short)0, acc1, false, false);

    // C/D layout: VGPR r, lane -> row M = r + 8*g, code col = m
    #pragma unroll
    for (int r = 0; r < 8; ++r) {
      float d0 = cn - 2.0f * acc0[r];
      float d1 = cn - 2.0f * acc1[r];
      bool l0 = d0 < bv0[r];                 // strict < keeps lowest index
      bool l1 = d1 < bv1[r];
      bv0[r] = l0 ? d0 : bv0[r];  bi0[r] = l0 ? code : bi0[r];
      bv1[r] = l1 ? d1 : bv1[r];  bi1[r] = l1 ? code : bi1[r];
    }
  }

  // ---- argmin reduce across the 16 lanes of each half-group ----
  #pragma unroll
  for (int r = 0; r < 8; ++r) {
    #pragma unroll
    for (int msk = 1; msk <= 8; msk <<= 1) {
      float ov0 = __shfl_xor(bv0[r], msk, 32);
      int   oi0 = __shfl_xor(bi0[r], msk, 32);
      if (ov0 < bv0[r] || (ov0 == bv0[r] && oi0 < bi0[r])) { bv0[r] = ov0; bi0[r] = oi0; }
      float ov1 = __shfl_xor(bv1[r], msk, 32);
      int   oi1 = __shfl_xor(bi1[r], msk, 32);
      if (ov1 < bv1[r] || (ov1 == bv1[r] && oi1 < bi1[r])) { bv1[r] = ov1; bi1[r] = oi1; }
    }
  }

  // ---- per-pair loss partial: sum(bestdist + ||z||^2) == sum ||z - e*||^2 ----
  float rowss = (sumsq0 + sumsq1) + __shfl_xor(sumsq0 + sumsq1, 16, 32);
  float bsum = 0.f;
  #pragma unroll
  for (int r = 0; r < 8; ++r) bsum += bv0[r] + bv1[r];
  float contrib = ((m == 0) ? bsum : 0.f) + ((lane < 16) ? rowss : 0.f);
  #pragma unroll
  for (int msk = 1; msk <= 16; msk <<= 1) contrib += __shfl_xor(contrib, msk, 32);
  if (lane == 0) partials[pair] = contrib;

  // ---- emit indices, usage flags, stash indices to LDS for the gather ----
  #pragma unroll
  for (int r = 0; r < 8; ++r) {
    if (m == r) {
      int row = 8 * g + r;
      idx_out[n0 + row]      = bi0[r];
      idx_out[n0 + 16 + row] = bi1[r];
      used[bi0[r]] = 1;               // idempotent stores: deterministic
      used[bi1[r]] = 1;
      sIdx[wave][row]      = bi0[r];
      sIdx[wave][16 + row] = bi1[r];
    }
  }
  __syncthreads();

  // ---- gather z_q = codebook[idx] and scatter back to [B,C,H,W] ----
  int idxA = sIdx[wave][m];                          // row m      -> w = w0 + m
  int idxB = sIdx[wave][16 + m];                     // row 16 + m -> w = w0 + 16 + m
  const float* crowA = codebook + (size_t)idxA * DIM;
  const float* crowB = codebook + (size_t)idxB * DIM;
  float* ob = zq_out + (size_t)b * CHW + hw0 + m;
  #pragma unroll 8
  for (int i = 0; i < 32; ++i) {
    int c = 32 * g + i;
    size_t off = (size_t)c * HW;
    ob[off]      = crowA[c];                         // 16-lane coalesced stores
    ob[off + 16] = crowB[c];
  }
}

// ---------------------------------------------------------------------------
// Finalize: deterministic tree reductions -> vq_loss, usage
// ---------------------------------------------------------------------------
__global__ __launch_bounds__(256) void vq_finalize(const float* __restrict__ partials,
                                                   const int* __restrict__ used,
                                                   float* __restrict__ out2) {
  __shared__ float sf[256];
  __shared__ int   si[256];
  int tid = threadIdx.x;
  float s = 0.f;
  for (int i = tid; i < NPAIRS; i += 256) s += partials[i];
  int c = 0;
  for (int i = tid; i < N_CODES; i += 256) c += (used[i] != 0) ? 1 : 0;
  sf[tid] = s; si[tid] = c;
  __syncthreads();
  for (int off = 128; off > 0; off >>= 1) {
    if (tid < off) { sf[tid] += sf[tid + off]; si[tid] += si[tid + off]; }
    __syncthreads();
  }
  if (tid == 0) {
    out2[0] = 1.25f * sf[0] / (float)NELEM;   // codebook_loss + 0.25*commit (same fwd value)
    out2[1] = (float)si[0] * (1.0f / (float)N_CODES);
  }
}

// ---------------------------------------------------------------------------
extern "C" void kernel_launch(void* const* d_in, const int* in_sizes, int n_in,
                              void* d_out, int out_size, void* d_ws, size_t ws_size,
                              hipStream_t stream) {
  const float* z        = (const float*)d_in[0];
  const float* codebook = (const float*)d_in[1];

  char* ws = (char*)d_ws;
  __bf16* cb_hi    = (__bf16*)(ws);               // 128 KB
  __bf16* cb_lo    = (__bf16*)(ws + 131072);      // 128 KB
  float*  cbnorm   = (float*)(ws + 262144);       // 4 KB
  int*    used     = (int*)(ws + 266240);         // 4 KB
  float*  partials = (float*)(ws + 270336);       // 16 KB

  // Output tuple order: z_q_st [8388608 f32], vq_loss, usage, indices [131072 i32]
  float* zq      = (float*)d_out;
  float* scalars = (float*)d_out + NELEM;
  int*   idx_out = (int*)d_out + (NELEM + 2);

  vq_prep    <<<4,          256, 0, stream>>>(codebook, cb_hi, cb_lo, cbnorm, used);
  vq_main    <<<NPAIRS / 8, 256, 0, stream>>>(z, codebook, cb_hi, cb_lo, cbnorm,
                                              zq, idx_out, used, partials);
  vq_finalize<<<1,          256, 0, stream>>>(partials, used, scalars);
}
